// Model_76699525972383
// MI455X (gfx1250) — matrix operand order repaired and checked by reading, not compiled, not run on hardware
//
#include <hip/hip_runtime.h>
#include <stdint.h>

// ---------------------------------------------------------------------------
// P-NET chain on MI455X (gfx1250, wave32):
//   diag (tanh per-gene)  -> bf16 A0
//   5x [ C = tanh(A @ bf16(W*M)^T-packed + b) ]  via v_wmma_f32_16x16x32_bf16
//   6x sigmoid heads -> d_out (4096 x 6) fp32
// GEMM: block tile 128x128 (8 waves), wave tile 32x64 (2x4 WMMA accums).
// K padded to 64 -> even number of 32-wide K-steps, so a ping-pong
// double-buffered loop (two named fragment sets, no register rotation moves)
// needs no tail branches and EXEC stays all-ones throughout.
// ---------------------------------------------------------------------------

typedef __attribute__((ext_vector_type(16))) __bf16 v16bf;
typedef __attribute__((ext_vector_type(8)))  __bf16 v8bf;
typedef __attribute__((ext_vector_type(8)))  float  v8f;

static __device__ __forceinline__ __bf16 f2bf(float f) {
  uint32_t u = __builtin_bit_cast(uint32_t, f);
  uint32_t r = u + 0x7FFFu + ((u >> 16) & 1u);   // round-to-nearest-even
  uint16_t h = (uint16_t)(r >> 16);
  return __builtin_bit_cast(__bf16, h);
}
static __device__ __forceinline__ float bf2f(__bf16 b) {
  uint32_t u = ((uint32_t)__builtin_bit_cast(uint16_t, b)) << 16;
  return __builtin_bit_cast(float, u);
}

// gfx1250 hardware TANH transcendental (confirmed: TRANS32 deps in asm).
static __device__ __forceinline__ float fast_tanh(float x) {
#if __has_builtin(__builtin_amdgcn_tanhf)
  return __builtin_amdgcn_tanhf(x);
#else
  return tanhf(x);
#endif
}

#define BATCH 4096

// ---------------------------------------------------------------------------
// Diagonal layer: A0[b,g] = tanh(sum_f x[b, 3g+f] * wd[g,f] + bd[g]), bf16 out,
// zero-filled K padding (g in [9229, 9280)). HBM-bound on the 453MB x stream.
// ---------------------------------------------------------------------------
__global__ void diag_kernel(const float* __restrict__ x,
                            const float* __restrict__ wd,
                            const float* __restrict__ bd,
                            __bf16* __restrict__ A0) {
  const int ld = 9280, G = 9229, XL = 27687;
  long long idx = (long long)blockIdx.x * blockDim.x + threadIdx.x;
  int b = (int)(idx / ld);
  int g = (int)(idx - (long long)b * ld);
  if (b >= BATCH) return;
  float v = 0.f;
  if (g < G) {
    const float* xr = x + (size_t)b * XL + 3 * g;
    const float* w  = wd + 3 * g;
    v = fast_tanh(xr[0] * w[0] + xr[1] * w[1] + xr[2] * w[2] + bd[g]);
  }
  A0[(size_t)b * ld + g] = f2bf(v);
}

// ---------------------------------------------------------------------------
// Pack: Wt[n, k] = bf16(W[k, n] * M[k, n]); transposed so B-fragments are
// contiguous-in-K per lane. Zero fill K-pad cols and N-pad rows (rows padded
// to a multiple of 128 so full block tiles never read OOB). LDS 32x32 tiles.
// ---------------------------------------------------------------------------
__global__ void pack_kernel(const float* __restrict__ W,
                            const float* __restrict__ Msk,
                            __bf16* __restrict__ Wt,
                            int din, int dout, int Kpad) {
  __shared__ float tile[32][33];
  int k0 = blockIdx.x * 32;
  int n0 = blockIdx.y * 32;
  int tx = threadIdx.x, ty = threadIdx.y;   // (32, 8)
  for (int r = 0; r < 4; ++r) {
    int k = k0 + ty + r * 8;
    int n = n0 + tx;
    float v = 0.f;
    if (k < din && n < dout) v = W[(size_t)k * dout + n] * Msk[(size_t)k * dout + n];
    tile[ty + r * 8][tx] = v;
  }
  __syncthreads();
  for (int r = 0; r < 4; ++r) {
    int n = n0 + ty + r * 8;
    int k = k0 + tx;
    Wt[(size_t)n * Kpad + k] = f2bf(tile[tx][ty + r * 8]);
  }
}

// ---------------------------------------------------------------------------
// bf16 WMMA GEMM + bias + tanh. A: (4096, K) bf16, Wt: (ceil128(N), K) bf16,
// C: (4096, ldC) bf16. Block = 8 waves = 128x128; wave = 32x64 (2x4 frags).
// Ping-pong K loop: 12 b128 loads of one fragment set fly while the other
// set's 8 WMMAs execute; K/32 is even so there are no tail branches.
// ---------------------------------------------------------------------------
__global__ __launch_bounds__(256) void gemm_tanh_kernel(
    const __bf16* __restrict__ A, const __bf16* __restrict__ Wt,
    const float* __restrict__ bias, __bf16* __restrict__ C,
    int K, int N, int ldC) {
  const int lane = threadIdx.x & 31;
  const int wave = threadIdx.x >> 5;
  const int wy = wave >> 1;                 // 0..3 -> 32-row slab
  const int wx = wave & 1;                  // 0..1 -> 64-col slab
  const int m0 = blockIdx.y * 128 + wy * 32;
  const int n0 = blockIdx.x * 128 + wx * 64;

  const int half  = lane >> 4;              // 0: lanes 0-15, 1: lanes 16-31
  const int mrow  = lane & 15;
  const int akoff = half ? 16 : 0;          // A frag: K base per half-wave
  const int bkoff = half ? 16 : 0;          // B frag: K base per half-wave

  v8f c0[4] = {v8f{}, v8f{}, v8f{}, v8f{}};
  v8f c1[4] = {v8f{}, v8f{}, v8f{}, v8f{}};

  const __bf16* arow0 = A + (size_t)(m0 + mrow) * K;
  const __bf16* arow1 = A + (size_t)(m0 + 16 + mrow) * K;
  const __bf16* brow[4];
  #pragma unroll
  for (int j = 0; j < 4; ++j)
    brow[j] = Wt + (size_t)(n0 + j * 16 + mrow) * K;   // column n0+16j+mrow

  // A 16x32 frag: elems [0..15] = K..K+15 contiguous (per half-wave).
  // B 32x16 frag: fixed column, 16 contiguous K values per lane.
  auto load_frags = [&](int k, v16bf& a0, v16bf& a1, v16bf b[4]) {
    v8bf a0lo = *(const v8bf*)(arow0 + k + akoff);
    v8bf a0hi = *(const v8bf*)(arow0 + k + akoff + 8);
    v8bf a1lo = *(const v8bf*)(arow1 + k + akoff);
    v8bf a1hi = *(const v8bf*)(arow1 + k + akoff + 8);
    a0 = __builtin_shufflevector(a0lo, a0hi, 0,1,2,3,4,5,6,7,8,9,10,11,12,13,14,15);
    a1 = __builtin_shufflevector(a1lo, a1hi, 0,1,2,3,4,5,6,7,8,9,10,11,12,13,14,15);
    #pragma unroll
    for (int j = 0; j < 4; ++j)
      b[j] = *(const v16bf*)(brow[j] + k + bkoff);
  };
  auto do_wmma = [&](const v16bf& a0, const v16bf& a1, const v16bf b[4]) {
    #pragma unroll
    for (int j = 0; j < 4; ++j) {
      c0[j] = __builtin_amdgcn_wmma_f32_16x16x32_bf16(false, a0, false, b[j], (short)0, c0[j], false, false);
      c1[j] = __builtin_amdgcn_wmma_f32_16x16x32_bf16(false, a1, false, b[j], (short)0, c1[j], false, false);
    }
  };

  // Ping-pong double buffer: sets P and Q, no rotation copies.
  v16bf pa0, pa1, pb[4];     // set P
  v16bf qa0, qa1, qb[4];     // set Q
  load_frags(0, pa0, pa1, pb);

  int k = 32;
  for (; k < K - 32; k += 64) {
    load_frags(k, qa0, qa1, qb);        // Q loads fly over P's WMMAs
    do_wmma(pa0, pa1, pb);
    load_frags(k + 32, pa0, pa1, pb);   // P loads fly over Q's WMMAs
    do_wmma(qa0, qa1, qb);
  }
  // K/32 is even: exactly one step (k == K-32) remains.
  load_frags(k, qa0, qa1, qb);
  do_wmma(pa0, pa1, pb);
  do_wmma(qa0, qa1, qb);

  // Epilogue. C frag layout: VGPR v -> row M = v + 8*half, lane -> col N=lane&15.
  auto store_tile = [&](const v8f& c, int mt, int nt) {
    int n = nt + mrow;
    if (n >= ldC) return;
    int mbase = mt + half * 8;
    if (n >= N) {                     // zero-filled pad columns (next layer K)
      __bf16 z = f2bf(0.f);
      #pragma unroll
      for (int v = 0; v < 8; ++v) C[(size_t)(mbase + v) * ldC + n] = z;
    } else {
      float bia = bias[n];
      #pragma unroll
      for (int v = 0; v < 8; ++v)
        C[(size_t)(mbase + v) * ldC + n] = f2bf(fast_tanh(c[v] + bia));
    }
  };
  #pragma unroll
  for (int j = 0; j < 4; ++j) {
    store_tile(c0[j], m0,      n0 + j * 16);
    store_tile(c1[j], m0 + 16, n0 + j * 16);
  }
}

// ---------------------------------------------------------------------------
// Sigmoid head: out[b, col] = sigmoid(act[b, :d] . wl + bl). One block per row.
// ---------------------------------------------------------------------------
__global__ void head_kernel(const __bf16* __restrict__ act, int d, int ld,
                            const float* __restrict__ wl,
                            const float* __restrict__ bl,
                            float* __restrict__ out, int col) {
  __shared__ float red[256];
  int b = blockIdx.x;
  const __bf16* row = act + (size_t)b * ld;
  float s = 0.f;
  for (int i = threadIdx.x; i < d; i += 256) s += bf2f(row[i]) * wl[i];
  red[threadIdx.x] = s;
  __syncthreads();
  for (int off = 128; off > 0; off >>= 1) {
    if (threadIdx.x < off) red[threadIdx.x] += red[threadIdx.x + off];
    __syncthreads();
  }
  if (threadIdx.x == 0) {
    float v = red[0] + bl[0];
    out[(size_t)b * 6 + col] = 1.f / (1.f + __expf(-v));
  }
}

// ---------------------------------------------------------------------------
extern "C" void kernel_launch(void* const* d_in, const int* in_sizes, int n_in,
                              void* d_out, int out_size, void* d_ws, size_t ws_size,
                              hipStream_t stream) {
  (void)in_sizes; (void)n_in; (void)out_size; (void)ws_size;
  const float* x  = (const float*)d_in[0];
  const float* wd = (const float*)d_in[1];
  const float* bd = (const float*)d_in[2];
  const float* Ws[5]; const float* Ms[5]; const float* bs[5];
  for (int i = 0; i < 5; ++i) {
    Ws[i] = (const float*)d_in[3 + i];
    Ms[i] = (const float*)d_in[8 + i];
    bs[i] = (const float*)d_in[13 + i];
  }
  const float* wls[6]; const float* bls[6];
  for (int i = 0; i < 6; ++i) {
    wls[i] = (const float*)d_in[18 + i];
    bls[i] = (const float*)d_in[24 + i];
  }
  float* out = (float*)d_out;

  static const int DIMS[6] = {9229, 1387, 1066, 447, 147, 26};
  static const int KP[6]   = {9280, 1408, 1088, 448, 192, 64};     // round64 (K pad)
  static const int NRW[5]  = {1408, 1152, 512, 256, 128};          // round128 (Wt rows)

  // Workspace carve (256B aligned): activations (bf16) + packed weights (bf16)
  uint8_t* ws = (uint8_t*)d_ws;
  size_t off = 0;
  auto carve = [&](size_t bytes) -> uint8_t* {
    uint8_t* p = ws + off;
    off += (bytes + 255) & ~(size_t)255;
    return p;
  };
  __bf16* Aact[6];
  for (int i = 0; i < 6; ++i) Aact[i] = (__bf16*)carve((size_t)BATCH * KP[i] * 2);
  __bf16* Wt[5];
  for (int i = 0; i < 5; ++i) Wt[i] = (__bf16*)carve((size_t)NRW[i] * KP[i] * 2);

  // 1) Pack masked weights, transposed + bf16 (independent of activations).
  for (int i = 0; i < 5; ++i) {
    dim3 grid(KP[i] / 32, NRW[i] / 32), blk(32, 8);
    pack_kernel<<<grid, blk, 0, stream>>>(Ws[i], Ms[i], Wt[i],
                                          DIMS[i], DIMS[i + 1], KP[i]);
  }

  // 2) Diagonal layer (streams the 453MB input; HBM-bound).
  {
    long long tot = (long long)BATCH * KP[0];
    diag_kernel<<<(unsigned)(tot / 256), 256, 0, stream>>>(x, wd, bd, Aact[0]);
  }
  head_kernel<<<BATCH, 256, 0, stream>>>(Aact[0], DIMS[0], KP[0], wls[0], bls[0], out, 0);

  // 3) WMMA GEMM chain + heads.
  for (int i = 0; i < 5; ++i) {
    int N = DIMS[i + 1];
    dim3 grid((N + 127) / 128, BATCH / 128), blk(256);
    gemm_tanh_kernel<<<grid, blk, 0, stream>>>(Aact[i], Wt[i], bs[i], Aact[i + 1],
                                               KP[i], N, KP[i + 1]);
    head_kernel<<<BATCH, 256, 0, stream>>>(Aact[i + 1], N, KP[i + 1],
                                           wls[i + 1], bls[i + 1], out, i + 1);
  }
}
